// RNNLayer_10505490006501
// MI455X (gfx1250) — matrix operand order repaired and checked by reading, not compiled, not run on hardware
//
#include <hip/hip_runtime.h>

// ---------------------------------------------------------------------------
// RNN layer for MI455X (gfx1250), wave32 + WMMA bf16 + async-LDS staging.
//   B=64, T=512, D_IN=D_H=D_OUT=1024
//   1) Xb = bf16(X); Wb = bf16(W)                          (elementwise)
//   2) Xh = Xb @ Whx^T + b_h  (bf16 out)                   (parallel GEMM)
//   3) scan: h_{t+1} = Xh_t + h_t @ Whh^T                  (16 persistent WGs,
//      Whh slice resident in 132KB LDS, device barrier per step)
//   4) O = Xb @ Wox^T + H @ Woh^T + b_o                    (parallel GEMM)
// GEMM: block 256 thr (8 waves), block tile 64x512, wave tile 64x64,
//       A tile double-buffered in LDS via GLOBAL_LOAD_ASYNC_TO_LDS_B128.
// ---------------------------------------------------------------------------

typedef __attribute__((ext_vector_type(16))) __bf16 v16bf;
typedef __attribute__((ext_vector_type(8)))  __bf16 v8bf;
typedef __attribute__((ext_vector_type(8)))  float  v8f;
typedef __attribute__((ext_vector_type(4)))  float  v4f;

#define DIM         1024
#define LDW         2048
#define BATCH       64
#define TLEN        512
#define MROWS       (BATCH * TLEN)   // 32768
#define SCAN_BLOCKS 16
#define APITCH      40               // LDS pitch (bf16) for 64x32 A tiles (bank-spread)
#define WPITCH      1032             // LDS pitch (bf16) for scan W slice (64 x 1024)
#define SCAN_SMEM   (64 * WPITCH * 2)

// D = A(16x32 bf16) * B(32x16 bf16) + C(f32)
static __device__ inline v8f wmma_bf16(v16bf a, v16bf b, v8f c) {
  return __builtin_amdgcn_wmma_f32_16x16x32_bf16(
      false, a, false, b, (short)0, c, false, false);
}

// A-fragment (16x32): lanes 0-15 hold K=0..15, lanes 16-31 hold
// K=16..31 of row M=(lane&15). Works for global or LDS pointers.
static __device__ inline v16bf load_a(const __bf16* __restrict__ rowPtr, int lane) {
  const int cs = ((lane >> 4) & 1) * 16;
  v8bf lo = *(const v8bf*)(rowPtr + cs);
  v8bf hi = *(const v8bf*)(rowPtr + cs + 8);
  v16bf a;
#pragma unroll
  for (int i = 0; i < 8; ++i) { a[i] = lo[i]; a[8 + i] = hi[i]; }
  return a;
}

// B-fragment (32x16) for C = A @ W^T: lanes 0-15 hold K=0..15, lanes 16-31
// hold K=16..31 of column N=(lane&15). wRowPtr -> W row (n0 + lane&15) col k.
static __device__ inline v16bf load_b(const __bf16* __restrict__ wRowPtr, int lane) {
  return *(const v16bf*)(wRowPtr + ((lane >> 4) & 1) * 16);
}

// Async copy of a 64x32 bf16 A tile (global -> LDS), one B128 per thread.
// ASYNCcnt-tracked; gtile points at tile (row stride = DIM elements).
static __device__ inline void async_a_tile(const __bf16* __restrict__ gtile,
                                           unsigned ldsBase, int tid) {
  const int row = tid >> 2;
  const int ch  = (tid & 3) * 8;
  unsigned laddr = ldsBase + (unsigned)(row * APITCH + ch) * 2u;
  unsigned long long gaddr =
      (unsigned long long)(uintptr_t)(gtile + (size_t)row * DIM + ch);
  asm volatile("global_load_async_to_lds_b128 %0, %1, off"
               :: "v"(laddr), "v"(gaddr) : "memory");
}
#define WAIT_ASYNC(n) asm volatile("s_wait_asynccnt " #n ::: "memory")

// ---------------------------------------------------------------------------
// Elementwise conversions
// ---------------------------------------------------------------------------
__global__ void convert_w_kernel(const float* __restrict__ Wh,
                                 const float* __restrict__ Wo,
                                 __bf16* __restrict__ Whb,
                                 __bf16* __restrict__ Wob, int n) {
  int i = blockIdx.x * blockDim.x + threadIdx.x;
  if (i < n) { Whb[i] = (__bf16)Wh[i]; Wob[i] = (__bf16)Wo[i]; }
}

__global__ void convert_x_kernel(const float* __restrict__ X,
                                 __bf16* __restrict__ Xb, int n4) {
  int i = blockIdx.x * blockDim.x + threadIdx.x;
  if (i < n4) {
    v4f x = *((const v4f*)X + i);
    __bf16* o = Xb + (size_t)i * 4;
    o[0] = (__bf16)x[0]; o[1] = (__bf16)x[1];
    o[2] = (__bf16)x[2]; o[3] = (__bf16)x[3];
  }
}

// ---------------------------------------------------------------------------
// One K=1024 phase of the GEMM: acc[mt][j] += A_tile @ Wcol^T
// A staged in LDS (double-buffered async), B read from global (L2-resident).
// ---------------------------------------------------------------------------
static __device__ inline void gemm_phase(const __bf16* __restrict__ Acol,  // A + mb*DIM
                                         const __bf16* __restrict__ Wcol,  // Wb + coloff
                                         __bf16* sA, unsigned sbase,
                                         int n0, int lane, int tid,
                                         v8f acc[4][4]) {
  const int r = lane & 15;
  const __bf16* wrow[4];
#pragma unroll
  for (int j = 0; j < 4; ++j) wrow[j] = Wcol + (size_t)(n0 + j * 16 + r) * LDW;

  const unsigned bufBytes = 64u * APITCH * 2u;
  async_a_tile(Acol, sbase, tid);                         // prologue: buf 0, k=0

#pragma unroll 2
  for (int k = 0; k < DIM; k += 32) {
    const int cur = (k >> 5) & 1;
    if (k + 32 < DIM) {
      async_a_tile(Acol + (k + 32), sbase + (unsigned)(cur ^ 1) * bufBytes, tid);
      WAIT_ASYNC(0x1);          // oldest (cur buffer) complete
    } else {
      WAIT_ASYNC(0x0);
    }
    __syncthreads();            // cur buffer visible to all waves

    v16bf bfr[4];
#pragma unroll
    for (int j = 0; j < 4; ++j) bfr[j] = load_b(wrow[j] + k, lane);
    const __bf16* base = sA + (size_t)cur * 64 * APITCH;
#pragma unroll
    for (int mt = 0; mt < 4; ++mt) {
      v16bf a = load_a(base + (size_t)(mt * 16 + r) * APITCH, lane);
#pragma unroll
      for (int j = 0; j < 4; ++j) acc[mt][j] = wmma_bf16(a, bfr[j], acc[mt][j]);
    }
    __syncthreads();            // all waves done reading cur before overwrite
  }
}

// C[m,n] = A0[m,:] @ Wb[n,0:1024]^T (+ A1[m,:] @ Wb[n,1024:2048]^T) + bias[n]
__global__ __launch_bounds__(256)
void gemm_kernel(const __bf16* __restrict__ A0,   // [M,1024] bf16
                 const __bf16* __restrict__ A1,   // [M,1024] bf16 or nullptr
                 const __bf16* __restrict__ Wb,   // [1024, LDW] bf16
                 const float* __restrict__ bias,  // [1024]
                 float* __restrict__ Cf,          // fp32 out (or nullptr)
                 __bf16* __restrict__ Cb)         // bf16 out (or nullptr)
{
  __shared__ __attribute__((aligned(32))) __bf16 sA[2 * 64 * APITCH];
  const int tid  = threadIdx.x;
  const int lane = tid & 31;
  const int wid  = tid >> 5;
  const int r    = lane & 15;
  const int rhi  = (lane >> 4) & 1;
  const int mb   = blockIdx.x * 64;                 // block rows
  const int n0   = blockIdx.y * 512 + wid * 64;     // wave cols
  const unsigned sbase = (unsigned)(uintptr_t)(void*)sA;

  v8f acc[4][4];
#pragma unroll
  for (int i = 0; i < 4; ++i)
#pragma unroll
    for (int j = 0; j < 4; ++j) acc[i][j] = (v8f)(0.0f);

  gemm_phase(A0 + (size_t)mb * DIM, Wb,       sA, sbase, n0, lane, tid, acc);
  if (A1 != nullptr)
    gemm_phase(A1 + (size_t)mb * DIM, Wb + DIM, sA, sbase, n0, lane, tid, acc);

  // C/D layout: VGPR v holds row (v + 8*rhi), col (lane&15)
#pragma unroll
  for (int mt = 0; mt < 4; ++mt) {
#pragma unroll
    for (int j = 0; j < 4; ++j) {
      const int col  = n0 + j * 16 + r;
      const float bv = bias[col];
      const size_t row0 = (size_t)(mb + mt * 16 + 8 * rhi);
#pragma unroll
      for (int v = 0; v < 8; ++v) {
        const float val = acc[mt][j][v] + bv;
        if (Cb) Cb[(row0 + v) * DIM + col] = (__bf16)val;
        else    Cf[(row0 + v) * DIM + col] = val;
      }
    }
  }
}

// ---------------------------------------------------------------------------
// Persistent scan: 16 WGs, each owns 64 h-columns; its W_hh slice (64x1024
// bf16, padded) lives in LDS for all 512 steps. Device barrier per step.
// ---------------------------------------------------------------------------
static __device__ inline void grid_barrier(unsigned* ctr, unsigned target) {
  __threadfence();
  __syncthreads();
  if (threadIdx.x == 0) {
    __hip_atomic_fetch_add(ctr, 1u, __ATOMIC_ACQ_REL, __HIP_MEMORY_SCOPE_AGENT);
    while (__hip_atomic_load(ctr, __ATOMIC_ACQUIRE, __HIP_MEMORY_SCOPE_AGENT) < target)
      __builtin_amdgcn_s_sleep(2);
  }
  __syncthreads();
  __threadfence();
}

__global__ __launch_bounds__(256)
void scan_kernel(const __bf16* __restrict__ Xhb,  // [B,T,D] bf16 (has b_h)
                 const __bf16* __restrict__ Whb,  // [1024, LDW] bf16
                 const float* __restrict__ h0,    // [B,D] fp32 initial hidden
                 __bf16* __restrict__ Hb,         // [B,T,D] bf16 entry hiddens
                 unsigned* __restrict__ ctr) {
  extern __shared__ __bf16 sW[];                  // [64][WPITCH]
  const int lane = threadIdx.x & 31;
  const int wid  = threadIdx.x >> 5;
  const int r    = lane & 15;
  const int rhi  = (lane >> 4) & 1;
  const int nbase = blockIdx.x * 64;

  // Stage W_hh slice: rows nbase..nbase+63, cols 1024..2047  ->  LDS
  for (int c = threadIdx.x; c < 64 * (DIM / 8); c += 256) {
    const int row = c >> 7;            // 128 8-elt chunks per row
    const int ch  = (c & 127) * 8;
    *(v8bf*)&sW[row * WPITCH + ch] =
        *(const v8bf*)(Whb + (size_t)(nbase + row) * LDW + DIM + ch);
  }

  // H[:,0,:] = h0 (column slice owned by this block)
  for (int idx = threadIdx.x; idx < BATCH * 64; idx += 256) {
    const int b = idx >> 6, c = idx & 63;
    Hb[((size_t)b * TLEN) * DIM + nbase + c] = (__bf16)h0[(size_t)b * DIM + nbase + c];
  }
  unsigned gen = 1;
  grid_barrier(ctr, gen * SCAN_BLOCKS);           // also covers LDS staging

  const int b0   = (wid & 3) * 16;                // wave's 16 batch rows
  const int nl0  = (wid >> 2) * 32;               // wave's 2 local col tiles
  const __bf16* swrow0 = sW + (size_t)(nl0 + r) * WPITCH;
  const __bf16* swrow1 = sW + (size_t)(nl0 + 16 + r) * WPITCH;

  for (int t = 0; t < TLEN - 1; ++t) {
    v8f acc0 = (v8f)(0.0f), acc1 = (v8f)(0.0f);
    const __bf16* hrow = Hb + ((size_t)(b0 + r) * TLEN + t) * DIM;
#pragma unroll 8
    for (int k = 0; k < DIM; k += 32) {
      v16bf a   = load_a(hrow + k, lane);
      v16bf bb0 = load_b(swrow0 + k, lane);
      v16bf bb1 = load_b(swrow1 + k, lane);
      acc0 = wmma_bf16(a, bb0, acc0);
      acc1 = wmma_bf16(a, bb1, acc1);
    }
    // h_{t+1} = acc + Xh_t (f32 add), store bf16 into H[:,t+1,:]
#pragma unroll
    for (int j = 0; j < 2; ++j) {
      const int col = nbase + nl0 + j * 16 + r;
      const v8f accj = j ? acc1 : acc0;
#pragma unroll
      for (int v = 0; v < 8; ++v) {
        const int brow = b0 + v + 8 * rhi;
        const float val = accj[v] +
            (float)Xhb[((size_t)brow * TLEN + t) * DIM + col];
        Hb[((size_t)brow * TLEN + (t + 1)) * DIM + col] = (__bf16)val;
      }
    }
    ++gen;
    grid_barrier(ctr, gen * SCAN_BLOCKS);
  }
}

// ---------------------------------------------------------------------------
// Launch
// ---------------------------------------------------------------------------
extern "C" void kernel_launch(void* const* d_in, const int* in_sizes, int n_in,
                              void* d_out, int out_size, void* d_ws, size_t ws_size,
                              hipStream_t stream) {
  const float* X  = (const float*)d_in[0];   // [B,T,D_IN]
  const float* h0 = (const float*)d_in[1];   // [B,D_H]
  const float* Wh = (const float*)d_in[2];   // [D_H, 2048]
  const float* bh = (const float*)d_in[3];   // [D_H]
  const float* Wo = (const float*)d_in[4];   // [D_OUT, 2048]
  const float* bo = (const float*)d_in[5];   // [D_OUT]
  float* out = (float*)d_out;                // [B,T,D_OUT]

  char* ws = (char*)d_ws;
  __bf16*  Whb = (__bf16*)(ws);                          //   4 MB
  __bf16*  Wob = (__bf16*)(ws + ((size_t)4   << 20));    //   4 MB
  __bf16*  Xb  = (__bf16*)(ws + ((size_t)8   << 20));    //  64 MB
  __bf16*  Xhb = (__bf16*)(ws + ((size_t)72  << 20));    //  64 MB
  __bf16*  Hb  = (__bf16*)(ws + ((size_t)136 << 20));    //  64 MB
  unsigned* ctr = (unsigned*)(ws + ((size_t)200 << 20)); //  barrier

  (void)hipFuncSetAttribute((const void*)scan_kernel,
                            hipFuncAttributeMaxDynamicSharedMemorySize, SCAN_SMEM);

  hipMemsetAsync(ctr, 0, 256, stream);

  const int nW = DIM * LDW;
  convert_w_kernel<<<(nW + 255) / 256, 256, 0, stream>>>(Wh, Wo, Whb, Wob, nW);
  const int n4 = MROWS * DIM / 4;
  convert_x_kernel<<<(n4 + 255) / 256, 256, 0, stream>>>(X, Xb, n4);

  // Xh = Xb @ Whx^T + b_h  (bf16 out)
  gemm_kernel<<<dim3(MROWS / 64, DIM / 512), 256, 0, stream>>>(
      Xb, nullptr, Whb, bh, nullptr, Xhb);

  // serial scan over T
  scan_kernel<<<SCAN_BLOCKS, 256, SCAN_SMEM, stream>>>(Xhb, Whb, h0, Hb, ctr);

  // O = Xb @ Wox^T + H @ Woh^T + b_o  (fp32 out)
  gemm_kernel<<<dim3(MROWS / 64, DIM / 512), 256, 0, stream>>>(
      Xb, Hb, Wob, bo, out, nullptr);
}
